// GPT_59579786330656
// MI455X (gfx1250) — matrix operand order repaired and checked by reading, not compiled, not run on hardware
//
#include <hip/hip_runtime.h>
#include <cstdint>
#include <cstddef>

// ---------------------------------------------------------------------------
// CDNA5 (gfx1250) GPT-2 forward. All GEMMs + attention matmuls use
// v_wmma_f32_16x16x32_bf16 (bf16 inputs, fp32 accumulate), wave32.
// GEMM staging uses GLOBAL_LOAD_ASYNC_TO_LDS_B128 + ASYNCcnt double buffering.
// ---------------------------------------------------------------------------

typedef __attribute__((ext_vector_type(16))) __bf16 v16bf;
typedef __attribute__((ext_vector_type(8)))  float  v8f;
typedef __attribute__((ext_vector_type(4)))  int    v4i;

union Frag { v16bf bf; uint4 q[2]; };

__device__ __forceinline__ unsigned short f32_to_bf16(float f) {
  unsigned u = __builtin_bit_cast(unsigned, f);
  u += 0x7FFFu + ((u >> 16) & 1u);           // round-to-nearest-even
  return (unsigned short)(u >> 16);
}

__device__ __forceinline__ v8f zero_v8f() {
  v8f z; for (int e = 0; e < 8; ++e) z[e] = 0.0f; return z;
}

__device__ __forceinline__ v8f wmma_bf16(v16bf a, v16bf b, v8f c) {
  // (neg_a, A, neg_b, B, c_mod, C, reuse_a, reuse_b)
  return __builtin_amdgcn_wmma_f32_16x16x32_bf16(false, a, false, b,
                                                 (short)0, c, false, false);
}

// ----------------------- async global -> LDS (16 bytes) --------------------
#if __has_builtin(__builtin_amdgcn_global_load_async_to_lds_b128)
#define HAVE_ASYNC_LDS 1
#endif

#if defined(HAVE_ASYNC_LDS)
#if __has_builtin(__builtin_amdgcn_s_wait_asynccnt)
#define ASYNC_WAIT(n) __builtin_amdgcn_s_wait_asynccnt(n)
#else
#define ASYNC_WAIT(n) asm volatile("s_wait_asynccnt %0" ::"i"(n) : "memory")
#endif
#else
#define ASYNC_WAIT(n)
#endif

typedef __attribute__((address_space(1))) v4i* as1_v4i_p;
typedef __attribute__((address_space(3))) v4i* as3_v4i_p;

__device__ __forceinline__ void async_copy16(const unsigned short* g, unsigned short* l) {
#if defined(HAVE_ASYNC_LDS)
  __builtin_amdgcn_global_load_async_to_lds_b128(
      (as1_v4i_p)(v4i*)(void*)g,
      (as3_v4i_p)(v4i*)(void*)l, 0, 0);
#else
  *(uint4*)l = *(const uint4*)g;
#endif
}

// A-matrix fragment, 16x32 bf16. Per CDNA5 ISA 7.12.2:
// lane row m = lane&15; elems 0..7 -> k = 8*half + e ; elems 8..15 -> k = 16 + 8*half + (e-8)
__device__ __forceinline__ v16bf load_afrag(const unsigned short* base, int pitch, int lane) {
  const int m = lane & 15, half = lane >> 4;
  Frag f;
  f.q[0] = *(const uint4*)(base + (size_t)m * pitch + 8 * half);
  f.q[1] = *(const uint4*)(base + (size_t)m * pitch + 16 + 8 * half);
  return f.bf;
}

// B-matrix fragment, 32x16 bf16, read from "transposed" storage where the row
// for column n holds its 32 contiguous k values: elems 0..7 -> k = 8*half + e ; elems 8..15 -> k = 16 + 8*half + (e-8).
__device__ __forceinline__ v16bf load_bfrag(const unsigned short* base, int pitch, int lane) {
  const int n = lane & 15, half = lane >> 4;
  Frag f;
  f.q[0] = *(const uint4*)(base + (size_t)n * pitch + 8 * half);
  f.q[1] = *(const uint4*)(base + (size_t)n * pitch + 16 + 8 * half);
  return f.bf;
}

// ------------------------------- model dims --------------------------------
constexpr int Cd = 768, Td = 2048, Bd = 2, Hd = 12, Ld = 4, Vd = 50257;
constexpr int C3 = 3 * Cd;        // 2304
constexpr int Fd = 4 * Cd;        // 3072
constexpr int Md = Bd * Td;       // 4096 rows of activations

// ------------------------------ small kernels ------------------------------

__global__ __launch_bounds__(256) void embed_k(const int* __restrict__ idx,
                                               const float* __restrict__ tok,
                                               const float* __restrict__ pos,
                                               float* __restrict__ x) {
  const int row = blockIdx.x;            // 0..Md-1
  const int t = row & (Td - 1);
  const int token = idx[row];
  for (int c = threadIdx.x; c < Cd; c += 256)
    x[(size_t)row * Cd + c] = tok[(size_t)token * Cd + c] + pos[(size_t)t * Cd + c];
}

__global__ __launch_bounds__(256) void layernorm_k(const float* __restrict__ x,
                                                   const float* __restrict__ g,
                                                   const float* __restrict__ beta,
                                                   unsigned short* __restrict__ out) {
  __shared__ float red[256];
  const int row = blockIdx.x, tid = threadIdx.x;
  const float* xr = x + (size_t)row * Cd;
  const float v0 = xr[tid], v1 = xr[tid + 256], v2 = xr[tid + 512];
  red[tid] = v0 + v1 + v2;
  __syncthreads();
  for (int off = 128; off; off >>= 1) {
    if (tid < off) red[tid] += red[tid + off];
    __syncthreads();
  }
  const float mean = red[0] * (1.0f / Cd);
  __syncthreads();
  const float d0 = v0 - mean, d1 = v1 - mean, d2 = v2 - mean;
  red[tid] = d0 * d0 + d1 * d1 + d2 * d2;
  __syncthreads();
  for (int off = 128; off; off >>= 1) {
    if (tid < off) red[tid] += red[tid + off];
    __syncthreads();
  }
  const float rinv = rsqrtf(red[0] * (1.0f / Cd) + 1e-5f);
  unsigned short* o = out + (size_t)row * Cd;
  o[tid]       = f32_to_bf16(d0 * rinv * g[tid]       + beta[tid]);
  o[tid + 256] = f32_to_bf16(d1 * rinv * g[tid + 256] + beta[tid + 256]);
  o[tid + 512] = f32_to_bf16(d2 * rinv * g[tid + 512] + beta[tid + 512]);
}

// W[K][N] fp32 (row-major) -> Wt[N][K] bf16
__global__ __launch_bounds__(256) void wtrans_k(const float* __restrict__ W,
                                                unsigned short* __restrict__ Wt,
                                                int K, int N) {
  const size_t i = (size_t)blockIdx.x * 256 + threadIdx.x;
  const size_t total = (size_t)K * N;
  if (i >= total) return;
  const int n = (int)(i % N);
  const int k = (int)(i / N);
  Wt[(size_t)n * K + k] = f32_to_bf16(W[i]);
}

// qkv bf16 [B,T,3C] -> Vt bf16 [B,H,64,T]
__global__ __launch_bounds__(256) void vtrans_k(const unsigned short* __restrict__ qkv,
                                                unsigned short* __restrict__ vt) {
  const size_t i = (size_t)blockIdx.x * 256 + threadIdx.x;   // < B*H*64*T
  const int t = (int)(i & (Td - 1));
  const size_t r = i >> 11;
  const int d = (int)(r & 63);
  const int bh = (int)(r >> 6);
  const int h = bh % Hd, b = bh / Hd;
  vt[i] = qkv[((size_t)b * Td + t) * C3 + 2 * Cd + h * 64 + d];
}

// ------------------------------- WMMA GEMM ---------------------------------
// C[M,N] = A[M,K](bf16) * Bt[N,K](bf16)^T + bias, optional residual / GELU.
// Block tile 128x128, 8 waves (4 row x 2 col), wave tile 32x64.
// Double-buffered LDS staging via GLOBAL_LOAD_ASYNC_TO_LDS_B128 (ASYNCcnt).
__global__ __launch_bounds__(256) void gemm_k(const unsigned short* __restrict__ A,
                                              const unsigned short* __restrict__ Bt,
                                              const float* __restrict__ bias,
                                              const float* __restrict__ res,
                                              float* __restrict__ outF,
                                              unsigned short* __restrict__ outB,
                                              int M, int N, int K, int gelu) {
  __shared__ unsigned short Al[2][128 * 40];
  __shared__ unsigned short Bl[2][128 * 40];
  const int tid = threadIdx.x;
  const int lane = tid & 31, wave = tid >> 5;
  const int wrow = wave & 3, wcol = wave >> 2;
  const int rowBase = blockIdx.y * 128;
  const int colBase = blockIdx.x * 128;

  // This thread's fixed staging assignment: 2 A chunks + 2 B chunks of 16B.
  const int r0 = tid >> 2;                  // 0..63
  const int r1 = r0 + 64;                   // 64..127
  const int roff = (tid & 3) << 3;          // 0,8,16,24 (bf16 elems)
  const unsigned short* ga0 = A + (size_t)(rowBase + r0) * K + roff;
  const unsigned short* ga1 = A + (size_t)(rowBase + r1) * K + roff;
  const int n0 = (colBase + r0 < N) ? (colBase + r0) : (N - 1);  // clamp keeps
  const int n1 = (colBase + r1 < N) ? (colBase + r1) : (N - 1);  // ASYNCcnt exact
  const unsigned short* gb0 = Bt + (size_t)n0 * K + roff;
  const unsigned short* gb1 = Bt + (size_t)n1 * K + roff;
  const int l0 = r0 * 40 + roff, l1 = r1 * 40 + roff;

  v8f acc[2][4];
  for (int i = 0; i < 2; ++i)
    for (int j = 0; j < 4; ++j) acc[i][j] = zero_v8f();

  // prologue: stage k-tile 0 into buffer 0
  async_copy16(ga0, &Al[0][l0]);
  async_copy16(ga1, &Al[0][l1]);
  async_copy16(gb0, &Bl[0][l0]);
  async_copy16(gb1, &Bl[0][l1]);

  int buf = 0;
  for (int kt = 0; kt < K; kt += 32) {
    __syncthreads();                 // previous compute done; buf^1 reusable
    if (kt + 32 < K) {
      const int nb = buf ^ 1;
      async_copy16(ga0 + kt + 32, &Al[nb][l0]);
      async_copy16(ga1 + kt + 32, &Al[nb][l1]);
      async_copy16(gb0 + kt + 32, &Bl[nb][l0]);
      async_copy16(gb1 + kt + 32, &Bl[nb][l1]);
      ASYNC_WAIT(4);                 // current buf landed (in-order), next in flight
    } else {
      ASYNC_WAIT(0);
    }
    __syncthreads();                 // all waves' current-buf loads visible

    const v16bf af0 = load_afrag(&Al[buf][(wrow * 32) * 40], 40, lane);
    const v16bf af1 = load_afrag(&Al[buf][(wrow * 32 + 16) * 40], 40, lane);
    for (int j = 0; j < 4; ++j) {
      const v16bf bfj = load_bfrag(&Bl[buf][(wcol * 64 + j * 16) * 40], 40, lane);
      acc[0][j] = wmma_bf16(af0, bfj, acc[0][j]);
      acc[1][j] = wmma_bf16(af1, bfj, acc[1][j]);
    }
    buf ^= 1;
  }

  const int n16 = lane & 15, half = lane >> 4;
  for (int i = 0; i < 2; ++i)
    for (int j = 0; j < 4; ++j) {
      const int cc = colBase + wcol * 64 + j * 16 + n16;
      if (cc >= N) continue;
      for (int v = 0; v < 8; ++v) {
        const int rr = rowBase + wrow * 32 + i * 16 + v + 8 * half;
        float val = acc[i][j][v];
        if (bias) val += bias[cc];
        if (res)  val += res[(size_t)rr * N + cc];
        if (gelu) val = 0.5f * val * (1.0f + erff(val * 0.70710678118f));
        if (outF) outF[(size_t)rr * N + cc] = val;
        if (outB) outB[(size_t)rr * N + cc] = f32_to_bf16(val);
      }
    }
}

// ------------------------- flash attention (causal) ------------------------
// One wave per 16 query rows; S=Q*K^T and O+=P*V via WMMA; online softmax.
__global__ __launch_bounds__(256) void attn_k(const unsigned short* __restrict__ qkv,
                                              const unsigned short* __restrict__ vt,
                                              unsigned short* __restrict__ y) {
  __shared__ unsigned short Pl[8 * 16 * 40];
  const int tid = threadIdx.x, lane = tid & 31, wave = tid >> 5;
  const int n16 = lane & 15, half = lane >> 4;
  const int h = blockIdx.y, b = blockIdx.z;
  const int qb = (blockIdx.x * 8 + wave) * 16;
  unsigned short* pl = &Pl[wave * 16 * 40];

  const unsigned short* qp = qkv + ((size_t)b * Td + qb) * C3 + h * 64;
  const v16bf qf0 = load_afrag(qp, C3, lane);
  const v16bf qf1 = load_afrag(qp + 32, C3, lane);

  float m8[8], l8[8];
  v8f o[4];
  for (int v = 0; v < 8; ++v) { m8[v] = -1e30f; l8[v] = 0.0f; }
  for (int j = 0; j < 4; ++j) o[j] = zero_v8f();

  const int kend = ((qb + 15) / 32 + 1) * 32;
  for (int kb = 0; kb < kend; kb += 32) {
    const unsigned short* kp = qkv + ((size_t)b * Td + kb) * C3 + Cd + h * 64;
    v8f s0 = zero_v8f(), s1 = zero_v8f();
    s0 = wmma_bf16(qf0, load_bfrag(kp, C3, lane), s0);
    s0 = wmma_bf16(qf1, load_bfrag(kp + 32, C3, lane), s0);
    s1 = wmma_bf16(qf0, load_bfrag(kp + (size_t)16 * C3, C3, lane), s1);
    s1 = wmma_bf16(qf1, load_bfrag(kp + (size_t)16 * C3 + 32, C3, lane), s1);

    for (int v = 0; v < 8; ++v) {
      const int qrow = qb + v + 8 * half;
      const int k0 = kb + n16, k1 = kb + 16 + n16;
      const bool ok0 = (k0 <= qrow), ok1 = (k1 <= qrow);
      const float a0 = ok0 ? s0[v] * 0.125f : -1e30f;   // 1/sqrt(64)
      const float a1 = ok1 ? s1[v] * 0.125f : -1e30f;
      float mx = fmaxf(a0, a1);
      for (int d = 8; d; d >>= 1) mx = fmaxf(mx, __shfl_xor(mx, d, 32));
      const float newm = fmaxf(m8[v], mx);
      const float corr = __expf(m8[v] - newm);
      const float p0 = ok0 ? __expf(a0 - newm) : 0.0f;
      const float p1 = ok1 ? __expf(a1 - newm) : 0.0f;
      float ps = p0 + p1;
      for (int d = 8; d; d >>= 1) ps += __shfl_xor(ps, d, 32);
      l8[v] = l8[v] * corr + ps;
      m8[v] = newm;
      for (int j = 0; j < 4; ++j) o[j][v] *= corr;
      pl[(v + 8 * half) * 40 + n16]      = f32_to_bf16(p0);
      pl[(v + 8 * half) * 40 + 16 + n16] = f32_to_bf16(p1);
    }

    const v16bf pf = load_afrag(pl, 40, lane);  // C-layout -> A-layout via LDS
    const unsigned short* vp = vt + ((size_t)(b * Hd + h) * 64) * Td + kb;
    for (int j = 0; j < 4; ++j)
      o[j] = wmma_bf16(pf, load_bfrag(vp + (size_t)(j * 16) * Td, Td, lane), o[j]);
  }

  for (int v = 0; v < 8; ++v) {
    const int qrow = qb + v + 8 * half;
    const float rl = 1.0f / l8[v];
    for (int j = 0; j < 4; ++j)
      y[((size_t)b * Td + qrow) * Cd + h * 64 + j * 16 + n16] =
          f32_to_bf16(o[j][v] * rl);
  }
}

// ------------------------------- launcher ----------------------------------

extern "C" void kernel_launch(void* const* d_in, const int* in_sizes, int n_in,
                              void* d_out, int out_size, void* d_ws, size_t ws_size,
                              hipStream_t stream) {
  (void)in_sizes; (void)n_in; (void)out_size; (void)ws_size;

  const int*   idx     = (const int*)  d_in[0];
  const float* tok_emb = (const float*)d_in[1];
  const float* pos_emb = (const float*)d_in[2];
  const float* ln1_g   = (const float*)d_in[3];
  const float* ln1_b   = (const float*)d_in[4];
  const float* qkv_w   = (const float*)d_in[5];
  const float* qkv_b   = (const float*)d_in[6];
  const float* aproj_w = (const float*)d_in[7];
  const float* aproj_b = (const float*)d_in[8];
  const float* ln2_g   = (const float*)d_in[9];
  const float* ln2_b   = (const float*)d_in[10];
  const float* fc_w    = (const float*)d_in[11];
  const float* fc_b    = (const float*)d_in[12];
  const float* mproj_w = (const float*)d_in[13];
  const float* mproj_b = (const float*)d_in[14];
  const float* lnf_g   = (const float*)d_in[15];
  const float* lnf_b   = (const float*)d_in[16];
  const float* head_w  = (const float*)d_in[17];
  float* logits = (float*)d_out;

  // workspace carve-up
  char* ws = (char*)d_ws;
  size_t off = 0;
  auto carve = [&](size_t bytes) -> void* {
    void* p = ws + off;
    off += (bytes + 255) & ~(size_t)255;
    return p;
  };
  float*          xf   = (float*)         carve((size_t)Md * Cd * 4);  // residual stream fp32
  unsigned short* hb   = (unsigned short*)carve((size_t)Md * Cd * 2);  // LN out bf16
  unsigned short* qkvb = (unsigned short*)carve((size_t)Md * C3 * 2);  // qkv bf16
  unsigned short* vtb  = (unsigned short*)carve((size_t)Bd * Hd * 64 * Td * 2);
  unsigned short* yb   = (unsigned short*)carve((size_t)Md * Cd * 2);  // attn out bf16
  unsigned short* midb = (unsigned short*)carve((size_t)Md * Fd * 2);  // MLP hidden bf16
  unsigned short* wt   = (unsigned short*)carve((size_t)Vd * Cd * 2);  // reusable Wt bf16

  auto gemm = [&](const unsigned short* A, const unsigned short* Bt,
                  const float* bias, const float* res, float* oF,
                  unsigned short* oB, int M, int N, int K, int gelu) {
    dim3 g((N + 127) / 128, M / 128);
    gemm_k<<<g, 256, 0, stream>>>(A, Bt, bias, res, oF, oB, M, N, K, gelu);
  };
  auto wtrans = [&](const float* W, int K, int N) {
    size_t total = (size_t)K * N;
    wtrans_k<<<dim3((unsigned)((total + 255) / 256)), 256, 0, stream>>>(W, wt, K, N);
  };

  // embeddings
  embed_k<<<Md, 256, 0, stream>>>(idx, tok_emb, pos_emb, xf);

  for (int l = 0; l < Ld; ++l) {
    // LN1 -> bf16
    layernorm_k<<<Md, 256, 0, stream>>>(xf, ln1_g + (size_t)l * Cd, ln1_b + (size_t)l * Cd, hb);
    // qkv = h @ qkv_w + qkv_b   (bf16 out)
    wtrans(qkv_w + (size_t)l * Cd * C3, Cd, C3);
    gemm(hb, wt, qkv_b + (size_t)l * C3, nullptr, nullptr, qkvb, Md, C3, Cd, 0);
    // V transpose for attention B-fragments
    {
      size_t total = (size_t)Bd * Hd * 64 * Td;
      vtrans_k<<<dim3((unsigned)(total / 256)), 256, 0, stream>>>(qkvb, vtb);
    }
    // causal flash attention -> y (bf16)
    attn_k<<<dim3(Td / 128, Hd, Bd), 256, 0, stream>>>(qkvb, vtb, yb);
    // x = x + y @ aproj_w + aproj_b   (fp32, in-place residual)
    wtrans(aproj_w + (size_t)l * Cd * Cd, Cd, Cd);
    gemm(yb, wt, aproj_b + (size_t)l * Cd, xf, xf, nullptr, Md, Cd, Cd, 0);
    // LN2 -> bf16
    layernorm_k<<<Md, 256, 0, stream>>>(xf, ln2_g + (size_t)l * Cd, ln2_b + (size_t)l * Cd, hb);
    // hid = gelu(h2 @ fc_w + fc_b)   (bf16 out)
    wtrans(fc_w + (size_t)l * Cd * Fd, Cd, Fd);
    gemm(hb, wt, fc_b + (size_t)l * Fd, nullptr, nullptr, midb, Md, Fd, Cd, 1);
    // x = x + hid @ mproj_w + mproj_b
    wtrans(mproj_w + (size_t)l * Fd * Cd, Fd, Cd);
    gemm(midb, wt, mproj_b + (size_t)l * Cd, xf, xf, nullptr, Md, Cd, Fd, 0);
  }

  // final LN + head
  layernorm_k<<<Md, 256, 0, stream>>>(xf, lnf_g, lnf_b, hb);
  wtrans(head_w, Cd, Vd);
  gemm(hb, wt, nullptr, nullptr, logits, nullptr, Md, Vd, Cd, 0);
}
